// AngleNet_81827716923454
// MI455X (gfx1250) — matrix operand order repaired and checked by reading, not compiled, not run on hardware
//
#include <hip/hip_runtime.h>
#include <cstdint>

// ---------------- problem constants (match reference) ----------------
#define N_ATOMS   500000
#define N_TYPES   16
#define N_FEAT    64
#define N_CENTER  40000
#define N_NEIGH   6
#define H1        128
#define H2        64
#define D_IN      195        // 3 + 3*64 (reference feature order)
#define KPAD      224        // padded K for f16 WMMA (7*32)
#define KS1       7          // K-steps layer 1 (224/32)
#define KS2       4          // K-steps layer 2 (128/32)
#define NT1       8          // N-tiles layer 1 (128/16)
#define NT2       4          // N-tiles layer 2 (64/16)
#define NPAIR     30         // nonzero off-diagonal (j,k) pairs
#define INV_CUT   (1.0f/3.5f)
#define PI_F      3.14159265358979f

#define WAVES     8
#define BLOCK     (WAVES*32)

typedef _Float16 v16h __attribute__((ext_vector_type(16)));
typedef _Float16 v8h  __attribute__((ext_vector_type(8)));
typedef _Float16 v4h  __attribute__((ext_vector_type(4)));
typedef float    v8f  __attribute__((ext_vector_type(8)));

// Permuted K layout (valid because the layer is linear in its inputs, and the
// weight packing below applies the same permutation):
//   k in [0,64)    : e_i[k]
//   k in [64,128)  : e_j[k-64]  / dij
//   k in [128,192) : e_k[k-128] / dik
//   k = 192,193,194: dij, dik, d_jk_norm
//   k = 195        : 1.0   (bias feature -> b1 folded into W1 column 195)
//   k in [196,224) : 0     (padding)
__device__ __forceinline__ float w1_elem(const float* __restrict__ W1,
                                         const float* __restrict__ b1,
                                         int n, int k) {
    if (k < 192)  return W1[n * D_IN + (k + 3)];
    if (k < 195)  return W1[n * D_IN + (k - 192)];
    if (k == 195) return b1[n];
    return 0.0f;
}

// ---------------------------------------------------------------------
// Pack W1/W2 into f16 wave32 B-fragments (B[k][n], 32x16 per fragment):
// lane half h carries N=lane%16; element s holds K = 8h+s (s<8) or
// K = 16+8h+(s-8) (s>=8), matching the A layout. Fragment idx is contiguous:
// halfword index (idx*32 + lane)*16 + 2v  ->  one 32B v16h per lane.
// ---------------------------------------------------------------------
__global__ void pack_w_kernel(const float* __restrict__ W1,
                              const float* __restrict__ b1,
                              const float* __restrict__ W2,
                              _Float16* __restrict__ w1p,
                              _Float16* __restrict__ w2p)
{
    const int tid    = blockIdx.x * blockDim.x + threadIdx.x;
    const int total1 = NT1 * KS1 * 32 * 8;     // one f16 pair per thread
    const int total2 = NT2 * KS2 * 32 * 8;
    if (tid < total1) {
        int v    = tid & 7;
        int lane = (tid >> 3) & 31;
        int ks   = (tid >> 8) % KS1;
        int nt   = (tid >> 8) / KS1;
        int n    = nt * 16 + (lane & 15);
        int k0   = ks * 32 + (lane >> 4) * 8 + (v >= 4 ? 8 : 0) + 2 * v;
        w1p[tid * 2]     = (_Float16)w1_elem(W1, b1, n, k0);
        w1p[tid * 2 + 1] = (_Float16)w1_elem(W1, b1, n, k0 + 1);
    } else if (tid < total1 + total2) {
        int t    = tid - total1;
        int v    = t & 7;
        int lane = (t >> 3) & 31;
        int ks   = (t >> 8) % KS2;
        int nt   = (t >> 8) / KS2;
        int n    = nt * 16 + (lane & 15);
        int k0   = ks * 32 + (lane >> 4) * 8 + (v >= 4 ? 8 : 0) + 2 * v;
        w2p[t * 2]     = (_Float16)W2[n * H1 + k0];
        w2p[t * 2 + 1] = (_Float16)W2[n * H1 + k0 + 1];
    }
}

__device__ __forceinline__ float celu1(float x) {
    return x > 0.0f ? x : (__expf(x) - 1.0f);
}

// ---------------------------------------------------------------------
// Main fused kernel: 1 wave == 1 center; 8 waves/block; no cross-wave
// sync (per-wave LDS regions; same-wave DS ops are in-order, ISA 7.3).
// ---------------------------------------------------------------------
__global__ __launch_bounds__(BLOCK) void anglenet_kernel(
    const float* __restrict__ atoms_xyz,
    const float* __restrict__ embed,
    const float* __restrict__ dist_ij,
    const float* __restrict__ b2,
    const int*   __restrict__ atom_type_idx,
    const int*   __restrict__ atom_i_idx,
    const int*   __restrict__ atom_j_idx,
    const _Float16* __restrict__ w1p,
    const _Float16* __restrict__ w2p,
    float* __restrict__ out)
{
    extern __shared__ char smem[];
    _Float16* Ast = (_Float16*)smem;                  // WAVES * 16*KPAD  f16
    _Float16* Hst = Ast + (size_t)WAVES * 16 * KPAD;  // WAVES * 16*H1    f16
    float*    Fcs = (float*)(Hst + (size_t)WAVES * 16 * H1);  // WAVES*16 f32

    const int lane = threadIdx.x & 31;
    const int wave = threadIdx.x >> 5;
    const int c    = blockIdx.x * WAVES + wave;
    if (c >= N_CENTER) return;

    _Float16* A   = Ast + wave * 16 * KPAD;
    _Float16* Hs  = Hst + wave * 16 * H1;
    float*    fcv = Fcs + wave * 16;

    const int m   = lane & 15;   // M-row within tile / N-col within C tile
    const int hlf = lane >> 4;   // lane half (selects K sub-group in A layout)

    // per-lane B-fragment base pointers (idx stride = 512 halfs = 1024 B)
    const _Float16* w1b = w1p + lane * 16;
    const _Float16* w2b = w2p + lane * 16;

    // warm WGP$/L2 with the packed weights
    __builtin_prefetch(w1b, 0, 3);
    __builtin_prefetch(w2b, 0, 3);

    // persistent descriptor accumulators: 4 N-tiles x 8 C-rows
    v8f dacc[NT2];
    #pragma unroll
    for (int i = 0; i < NT2; ++i)
        #pragma unroll
        for (int j = 0; j < 8; ++j) dacc[i][j] = 0.0f;

    #pragma unroll 1
    for (int half = 0; half < 2; ++half) {
        // ---------------- stage features (f16) into LDS ----------------
        const int  r     = m;               // both lane-halves serve row r
        const int  t     = half * 16 + r;   // pair index
        const int  p     = hlf;             // feature-slice selector (32 feats)
        const bool valid = (t < NPAIR);

        float dij = 0.f, dik = 0.f, djkn = 0.f, fc = 0.f;
        int ti = 0, tj = 0, tk = 0;
        if (valid) {
            int jj = t / 5;
            int q  = t - jj * 5;
            int kk = q + (q >= jj ? 1 : 0);
            int ai = atom_i_idx[c];
            int aj = atom_j_idx[c * N_NEIGH + jj];
            int ak = atom_j_idx[c * N_NEIGH + kk];
            float d0 = atoms_xyz[aj * 3 + 0] - atoms_xyz[ak * 3 + 0];
            float d1 = atoms_xyz[aj * 3 + 1] - atoms_xyz[ak * 3 + 1];
            float d2 = atoms_xyz[aj * 3 + 2] - atoms_xyz[ak * 3 + 2];
            float djk = __fsqrt_rn(d0 * d0 + d1 * d1 + d2 * d2);
            dij = dist_ij[c * N_NEIGH + jj];
            dik = dist_ij[c * N_NEIGH + kk];
            float mx = fmaxf(dij, dik), mn = fminf(dij, dik);
            djkn = (djk - mx + mn) / (2.0f * mn);
            ti = atom_type_idx[ai];
            tj = atom_type_idx[aj];
            tk = atom_type_idx[ak];
            float u = fminf(fmaxf(dij * INV_CUT, 0.0f), 1.0f);
            float w = fminf(fmaxf(dik * INV_CUT, 0.0f), 1.0f);
            fc = (0.5f * __cosf(PI_F * u) + 0.5f) *
                 (0.5f * __cosf(PI_F * w) + 0.5f);
        }
        _Float16* row = A + r * KPAD;

        // vectorized embedding staging: float4 loads, 4xf16 (b64) LDS stores
        const float sc0 = valid ? 1.0f : 0.0f;
        const float sc1 = valid ? 1.0f / dij : 0.0f;
        const float sc2 = valid ? 1.0f / dik : 0.0f;
        const float* eb0 = embed + ti * N_FEAT;
        const float* eb1 = embed + tj * N_FEAT;
        const float* eb2 = embed + tk * N_FEAT;
        #pragma unroll
        for (int seg = 0; seg < 3; ++seg) {
            const float* eb = (seg == 0) ? eb0 : (seg == 1) ? eb1 : eb2;
            const float  s  = (seg == 0) ? sc0 : (seg == 1) ? sc1 : sc2;
            const float4* src = (const float4*)(eb) + p * 8;
            _Float16*     dst = row + seg * 64 + p * 32;
            #pragma unroll
            for (int ch = 0; ch < 8; ++ch) {
                float4 a = src[ch];
                v4h hh;
                hh[0] = (_Float16)(a.x * s);
                hh[1] = (_Float16)(a.y * s);
                hh[2] = (_Float16)(a.z * s);
                hh[3] = (_Float16)(a.w * s);
                *(v4h*)(dst + ch * 4) = hh;          // ds_store_b64
            }
        }
        if (p == 0) {
            v4h hh;
            hh[0] = (_Float16)dij;
            hh[1] = (_Float16)dik;
            hh[2] = (_Float16)djkn;
            hh[3] = (_Float16)1.0f;                  // bias feature (k=195)
            *(v4h*)(row + 192) = hh;
            fcv[r] = fc;
        } else {
            v4h z = {};
            #pragma unroll
            for (int kq = 196; kq < KPAD; kq += 4)   // zero K padding
                *(v4h*)(row + kq) = z;
        }

        // preload this lane-half's 8 cutoff factors (rows v + 8*hlf)
        float fcr[8];
        #pragma unroll
        for (int v = 0; v < 8; ++v) fcr[v] = fcv[v + 8 * hlf];

        // ------- hoist all layer-1 A-fragments into registers (56 VGPRs) ---
        // A-frag (16-bit 16x32 layout): lane m = row, K sub-group by lane half
        v16h a1[KS1];
        #pragma unroll
        for (int ks = 0; ks < KS1; ++ks) {
            const v8h alo = *(const v8h*)(A + m * KPAD + ks * 32 + hlf * 8);
            const v8h ahi = *(const v8h*)(A + m * KPAD + ks * 32 + 16 + hlf * 8);
            #pragma unroll
            for (int j = 0; j < 8; ++j) { a1[ks][j] = alo[j]; a1[ks][8 + j] = ahi[j]; }
        }

        // ---------------- layer 1: 16x224 @ 224x128, WMMA ----------------
        // nt loop kept rolled (compact, low pressure); ks loop unrolled so B
        // loads get immediate offsets and the scheduler pipelines them.
        #pragma unroll 1
        for (int nt = 0; nt < NT1; ++nt) {
            v8f acc;
            #pragma unroll
            for (int j = 0; j < 8; ++j) acc[j] = 0.0f;
            #pragma unroll
            for (int ks = 0; ks < KS1; ++ks) {
                const v16h bv = *(const v16h*)(w1b + (nt * KS1 + ks) * 512);
                acc = __builtin_amdgcn_wmma_f32_16x16x32_f16(
                          false, a1[ks], false, bv, (short)0, acc, false, false);
            }
            // C layout: acc[v] holds M = v + 8*hlf, N = m (within tile)
            const int n = nt * 16 + m;
            #pragma unroll
            for (int v = 0; v < 8; ++v) {
                float x = celu1(acc[v]);             // b1 folded into GEMM
                Hs[(v + 8 * hlf) * H1 + n] = (_Float16)x;
            }
        }

        // ------- hoist layer-2 A-fragments into registers (32 VGPRs) -------
        v16h a2[KS2];
        #pragma unroll
        for (int ks = 0; ks < KS2; ++ks) {
            const v8h alo = *(const v8h*)(Hs + m * H1 + ks * 32 + hlf * 8);
            const v8h ahi = *(const v8h*)(Hs + m * H1 + ks * 32 + 16 + hlf * 8);
            #pragma unroll
            for (int j = 0; j < 8; ++j) { a2[ks][j] = alo[j]; a2[ks][8 + j] = ahi[j]; }
        }

        // ------- layer 2: 16x128 @ 128x64, WMMA + cutoff accumulate -------
        #pragma unroll 1
        for (int nt = 0; nt < NT2; ++nt) {
            v8f acc;
            #pragma unroll
            for (int j = 0; j < 8; ++j) acc[j] = 0.0f;
            #pragma unroll
            for (int ks = 0; ks < KS2; ++ks) {
                const v16h bv = *(const v16h*)(w2b + (nt * KS2 + ks) * 512);
                acc = __builtin_amdgcn_wmma_f32_16x16x32_f16(
                          false, a2[ks], false, bv, (short)0, acc, false, false);
            }
            const int   n    = nt * 16 + m;
            const float bias = b2[n];
            #pragma unroll
            for (int v = 0; v < 8; ++v) {
                float x = celu1(acc[v] + bias);
                dacc[nt][v] += x * fcr[v];           // fc==0 kills pad rows
            }
        }
    }

    // ---- reduce over the 16 M-rows (pairs) and write desc[c, :] ----
    #pragma unroll
    for (int nt = 0; nt < NT2; ++nt) {
        float s = 0.0f;
        #pragma unroll
        for (int v = 0; v < 8; ++v) s += dacc[nt][v];
        s += __shfl_xor(s, 16, 32);      // combine lane halves (M 0-7 vs 8-15)
        if (hlf == 0)
            out[c * H2 + nt * 16 + m] = s;
    }
}

// ---------------------------------------------------------------------
extern "C" void kernel_launch(void* const* d_in, const int* in_sizes, int n_in,
                              void* d_out, int out_size, void* d_ws, size_t ws_size,
                              hipStream_t stream) {
    const float* atoms_xyz = (const float*)d_in[0];
    const float* embed     = (const float*)d_in[1];
    const float* dist_ij   = (const float*)d_in[2];
    const float* W1        = (const float*)d_in[3];
    const float* b1        = (const float*)d_in[4];
    const float* W2        = (const float*)d_in[5];
    const float* b2        = (const float*)d_in[6];
    const int*   atype     = (const int*)d_in[7];
    const int*   ai        = (const int*)d_in[8];
    const int*   aj        = (const int*)d_in[9];

    _Float16* w1p = (_Float16*)d_ws;                       // 57,344 B
    _Float16* w2p = w1p + (size_t)NT1 * KS1 * 32 * 16;     // 16,384 B

    const int packThreads = (NT1 * KS1 + NT2 * KS2) * 32 * 8;
    pack_w_kernel<<<(packThreads + 255) / 256, 256, 0, stream>>>(W1, b1, W2, w1p, w2p);

    const size_t smemBytes =
        (size_t)WAVES * 16 * KPAD * sizeof(_Float16) +     // A staging
        (size_t)WAVES * 16 * H1   * sizeof(_Float16) +     // H1 activations
        (size_t)WAVES * 16        * sizeof(float);         // cutoff factors

    const int blocks = (N_CENTER + WAVES - 1) / WAVES;
    anglenet_kernel<<<blocks, BLOCK, smemBytes, stream>>>(
        atoms_xyz, embed, dist_ij, b2, atype, ai, aj, w1p, w2p, (float*)d_out);
}
